// TransformerBlock_3934190044168
// MI455X (gfx1250) — matrix operand order repaired and checked by reading, not compiled, not run on hardware
//
#include <hip/hip_runtime.h>
#include <hip/hip_bf16.h>
#include <math.h>

typedef __attribute__((ext_vector_type(16))) _Float16 v16h;
typedef __attribute__((ext_vector_type(8)))  _Float16 v8h;
typedef __attribute__((ext_vector_type(8)))  float    v8f;

#define N_PTS  2048
#define M_NBR  16
#define H_HEADS 8
#define KA_PAD 224   // attention msg features 194 -> padded to 7*32

__device__ __forceinline__ float gelu_f(float x) {
    // exact GELU (approximate=False)
    return 0.5f * x * (1.0f + erff(x * 0.70710678f));
}

__device__ __forceinline__ v16h cat16(v8h lo, v8h hi) {
    return __builtin_shufflevector(lo, hi, 0, 1, 2, 3, 4, 5, 6, 7,
                                           8, 9, 10, 11, 12, 13, 14, 15);
}

// ---------------------------------------------------------------------------
// Generic WMMA GEMM (LDS-free, no guards): C[rows x Nout] = A[rows x K] @ Bt^T
// A row-major f16 (K contiguous), Bt transposed f16 [Nout][K] (K contiguous).
// block = 128 threads (4 waves); block tile = 32 rows x 128 cols.
// Each wave: 2 row-tiles x 2 col-tiles = 4 accumulators, 4 WMMA / 8 b128 loads
// per k-chunk.  Requires: rows % 32 == 0, Nout % 128 == 0 (pad), K % 32 == 0.
// act: 0 = none, 1 = GELU.  outF (f32) and/or outH (f16) may be null.
// ---------------------------------------------------------------------------
__global__ __launch_bounds__(128) void gemm_f16_kernel(
    const _Float16* __restrict__ A, const _Float16* __restrict__ Bt,
    const float* __restrict__ bias, float* __restrict__ outF,
    _Float16* __restrict__ outH, int K, int Nout, int act)
{
    const int tid  = threadIdx.x;
    const int wave = tid >> 5;
    const int lane = tid & 31;
    const int hi   = lane >> 4;
    const int l16  = lane & 15;
    const int row0 = blockIdx.x * 32;
    const int cb0  = blockIdx.y * 128 + (wave << 5);   // wave's first col tile
    const int cb1  = cb0 + 16;

    const _Float16* ar0 = A + (size_t)(row0 + l16) * K + (hi << 3);
    const _Float16* ar1 = ar0 + (size_t)16 * K;
    const _Float16* bp0 = Bt + (size_t)(cb0 + l16) * K + (hi << 3);
    const _Float16* bp1 = Bt + (size_t)(cb1 + l16) * K + (hi << 3);

    const v8f zero = {0.f, 0.f, 0.f, 0.f, 0.f, 0.f, 0.f, 0.f};
    v8f a00 = zero, a01 = zero, a10 = zero, a11 = zero;

    const int nk = K >> 5;
    for (int kc = 0; kc < nk; ++kc) {
        const int kb = kc << 5;
        const v16h fa0 = cat16(*(const v8h*)(ar0 + kb), *(const v8h*)(ar0 + kb + 16));
        const v16h fa1 = cat16(*(const v8h*)(ar1 + kb), *(const v8h*)(ar1 + kb + 16));
        const v16h fb0 = cat16(*(const v8h*)(bp0 + kb), *(const v8h*)(bp0 + kb + 16));
        const v16h fb1 = cat16(*(const v8h*)(bp1 + kb), *(const v8h*)(bp1 + kb + 16));
        a00 = __builtin_amdgcn_wmma_f32_16x16x32_f16(false, fa0, false, fb0, (short)0, a00, false, false);
        a01 = __builtin_amdgcn_wmma_f32_16x16x32_f16(false, fa0, false, fb1, (short)0, a01, false, false);
        a10 = __builtin_amdgcn_wmma_f32_16x16x32_f16(false, fa1, false, fb0, (short)0, a10, false, false);
        a11 = __builtin_amdgcn_wmma_f32_16x16x32_f16(false, fa1, false, fb1, (short)0, a11, false, false);
    }

#pragma unroll
    for (int rt = 0; rt < 2; ++rt) {
#pragma unroll
        for (int ct = 0; ct < 2; ++ct) {
            const v8f acc = rt ? (ct ? a11 : a10) : (ct ? a01 : a00);
            const int colg = (ct ? cb1 : cb0) + l16;
            const int rbase = row0 + (rt << 4) + (hi ? 8 : 0);
            const float bv = bias ? bias[colg] : 0.0f;
#pragma unroll
            for (int r = 0; r < 8; ++r) {
                float v = acc[r] + bv;
                if (act == 1) v = gelu_f(v);
                if (outF) outF[(size_t)(rbase + r) * Nout + colg] = v;
                if (outH) outH[(size_t)(rbase + r) * Nout + colg] = (_Float16)v;
            }
        }
    }
}

// ---------------------------------------------------------------------------
// Row LayerNorm (eps=1e-5, no affine): f32 in -> normalized f16 out
// ---------------------------------------------------------------------------
__global__ __launch_bounds__(256) void ln_f32_to_f16_kernel(
    const float* __restrict__ x, _Float16* __restrict__ out, int C)
{
    __shared__ float r1[256];
    __shared__ float r2[256];
    const int row = blockIdx.x;
    const int t = threadIdx.x;
    const float* xr = x + (size_t)row * C;
    float s1 = 0.f, s2 = 0.f;
    for (int c = t; c < C; c += 256) { const float v = xr[c]; s1 += v; s2 += v * v; }
    r1[t] = s1; r2[t] = s2;
    __syncthreads();
    for (int s = 128; s > 0; s >>= 1) {
        if (t < s) { r1[t] += r1[t + s]; r2[t] += r2[t + s]; }
        __syncthreads();
    }
    const float invC = 1.0f / (float)C;
    const float mean = r1[0] * invC;
    const float var  = r2[0] * invC - mean * mean;
    const float rstd = rsqrtf(var + 1e-5f);
    _Float16* orow = out + (size_t)row * C;
    for (int c = t; c < C; c += 256) orow[c] = (_Float16)((xr[c] - mean) * rstd);
}

// ---------------------------------------------------------------------------
// weight f32 [srcK x srcN] -> TRANSPOSED f16 [dstN x dstK] with zero padding
// ---------------------------------------------------------------------------
__global__ void convert_transpose_kernel(const float* __restrict__ src, _Float16* __restrict__ dst,
                                         int srcK, int srcN, int dstK, int dstN)
{
    const int i = blockIdx.x * blockDim.x + threadIdx.x;
    if (i >= dstN * dstK) return;
    const int n = i / dstK, k = i - n * dstK;
    dst[i] = (_Float16)((k < srcK && n < srcN) ? src[(size_t)k * srcN + n] : 0.f);
}

__global__ void pad_bias_kernel(const float* __restrict__ src, float* __restrict__ dst,
                                int nsrc, int ndst)
{
    const int i = blockIdx.x * blockDim.x + threadIdx.x;
    if (i < ndst) dst[i] = (i < nsrc) ? src[i] : 0.f;
}

__global__ void split_qkv_kernel(const float* __restrict__ qkv, _Float16* __restrict__ qh,
                                 _Float16* __restrict__ kh, float* __restrict__ vf)
{
    const int i = blockIdx.x * blockDim.x + threadIdx.x;
    if (i >= N_PTS * 512) return;
    const int n = i >> 9, d = i & 511;
    const float* row = qkv + (size_t)n * 1536;
    qh[i] = (_Float16)row[d];
    kh[i] = (_Float16)row[512 + d];
    vf[i] = row[1024 + d];
}

__global__ void add2_kernel(const float* __restrict__ a, const float* __restrict__ b,
                            float* __restrict__ out32, _Float16* __restrict__ out16, int n)
{
    const int i = blockIdx.x * blockDim.x + threadIdx.x;
    if (i >= n) return;
    const float v = a[i] + b[i];
    out32[i] = v;
    out16[i] = (_Float16)v;
}

__global__ void gd_kernel(const float* __restrict__ gene_exp, const int* __restrict__ nbr,
                          _Float16* __restrict__ gdh)
{
    const int i = blockIdx.x * blockDim.x + threadIdx.x;
    if (i >= N_PTS * M_NBR * 50) return;
    const int nm = i / 50, g = i - nm * 50;
    const int n = nm >> 4;
    gdh[i] = (_Float16)(gene_exp[(size_t)nbr[nm] * 50 + g] - gene_exp[(size_t)n * 50 + g]);
}

__global__ void copy_out_kernel(const float* __restrict__ g2, float* __restrict__ out)
{
    const int i = blockIdx.x * blockDim.x + threadIdx.x;
    if (i >= N_PTS * 50) return;
    const int n = i / 50, g = i - n * 50;
    out[i] = g2[(size_t)n * 128 + g];
}

// ---------------------------------------------------------------------------
// Frame averaging geometry: radial, centered cov, closed-form 2x2 eigh (ascending),
// projection into the eigenbasis, radial norm.  One thread per point n.
// ---------------------------------------------------------------------------
__global__ void frames_kernel(const float* __restrict__ coords, const int* __restrict__ nbr,
                              float* __restrict__ proj, float* __restrict__ rnorm)
{
    const int n = blockIdx.x * blockDim.x + threadIdx.x;
    if (n >= N_PTS) return;
    const float cx = coords[n * 2 + 0];
    const float cy = coords[n * 2 + 1];
    float rx[16], ry[16];
    float mx = 0.f, my = 0.f;
#pragma unroll
    for (int m = 0; m < 16; ++m) {
        const int id = nbr[n * 16 + m];
        rx[m] = coords[id * 2 + 0] - cx;
        ry[m] = coords[id * 2 + 1] - cy;
        mx += rx[m]; my += ry[m];
    }
    mx *= (1.f / 16.f); my *= (1.f / 16.f);
    float a = 0.f, b = 0.f, c2 = 0.f;
#pragma unroll
    for (int m = 0; m < 16; ++m) {
        const float x = rx[m] - mx, y = ry[m] - my;
        a += x * x; b += x * y; c2 += y * y;
    }
    float V00, V01, V10, V11;
    if (fabsf(b) < 1e-12f * (fabsf(a) + fabsf(c2)) + 1e-30f) {
        if (a <= c2) { V00 = 1.f; V10 = 0.f; V01 = 0.f; V11 = 1.f; }
        else         { V00 = 0.f; V10 = 1.f; V01 = 1.f; V11 = 0.f; }
    } else {
        const float dd = 0.5f * (a - c2);
        const float s  = sqrtf(dd * dd + b * b);
        const float l0 = 0.5f * (a + c2) - s;          // smaller eigenvalue (ascending)
        float vx = b, vy = l0 - a;
        const float inv = rsqrtf(vx * vx + vy * vy);
        vx *= inv; vy *= inv;
        V00 = vx; V10 = vy; V01 = -vy; V11 = vx;       // second column orthogonal
    }
#pragma unroll
    for (int m = 0; m < 16; ++m) {
        const float x = rx[m] - mx, y = ry[m] - my;
        proj[(size_t)(n * 16 + m) * 2 + 0] = x * V00 + y * V10;
        proj[(size_t)(n * 16 + m) * 2 + 1] = x * V01 + y * V11;
        rnorm[n * 16 + m] = sqrtf(rx[m] * rx[m] + ry[m] * ry[m]);
    }
}

// ---------------------------------------------------------------------------
// Edge MLP fc1 + GELU + LN, averaged over 4 sign frames.  Block = (n,m), 128 thr.
// (fc2 is linear and applied after the frame mean by the generic GEMM.)
// ---------------------------------------------------------------------------
__global__ __launch_bounds__(128) void edge_hidden_kernel(
    const float* __restrict__ proj, const float* __restrict__ rnorm,
    const float* __restrict__ we1, const float* __restrict__ be1,
    _Float16* __restrict__ hn_avg)
{
    __shared__ float r1[128];
    __shared__ float r2[128];
    const int nm = blockIdx.x;
    const int c = threadIdx.x;
    const float px = proj[(size_t)nm * 2 + 0];
    const float py = proj[(size_t)nm * 2 + 1];
    const float rn = rnorm[nm];
    const float w0 = we1[c], w1 = we1[128 + c], w2 = we1[256 + c], bb = be1[c];
    const float sx[4] = {-1.f, -1.f, 1.f, 1.f};
    const float sy[4] = {-1.f,  1.f, -1.f, 1.f};
    float acc = 0.f;
#pragma unroll
    for (int f = 0; f < 4; ++f) {
        const float g = gelu_f(sx[f] * px * w0 + sy[f] * py * w1 + rn * w2 + bb);
        r1[c] = g; r2[c] = g * g;
        __syncthreads();
        for (int s = 64; s > 0; s >>= 1) {
            if (c < s) { r1[c] += r1[c + s]; r2[c] += r2[c + s]; }
            __syncthreads();
        }
        const float mean = r1[0] * (1.f / 128.f);
        const float var  = r2[0] * (1.f / 128.f) - mean * mean;
        acc += (g - mean) * rsqrtf(var + 1e-5f);
        __syncthreads();
    }
    hn_avg[(size_t)nm * 128 + c] = (_Float16)(acc * 0.25f);
}

// ---------------------------------------------------------------------------
// Attention MLP: gather msg panel (q|k[nbr]|edge|gd, padded 194->224) into LDS,
// WMMA fc1 (224->512) against transposed weights (direct global b128 fragments),
// bias+GELU, parallel LN over 512, dot with wa2 -> logit per edge-head.
// Block = 16 edge-head rows, 128 threads (4 waves x 8 col tiles).
// ---------------------------------------------------------------------------
__global__ __launch_bounds__(128) void attn_fc_kernel(
    const _Float16* __restrict__ qh, const _Float16* __restrict__ kh,
    const _Float16* __restrict__ edgeh, const _Float16* __restrict__ gdh,
    const int* __restrict__ nbr, const _Float16* __restrict__ wa1t,  // [512][224]
    const float* __restrict__ ba1, const float* __restrict__ wa2,
    const float* __restrict__ ba2, float* __restrict__ logits)
{
    __shared__ __attribute__((aligned(16))) _Float16 Ap[16][KA_PAD];
    __shared__ float Hb[16][512];
    __shared__ float red1[16][8];
    __shared__ float red2[16][8];

    const int tid  = threadIdx.x;
    const int row0 = blockIdx.x * 16;          // global edge-head row
    const int n    = row0 >> 7;                // 128 rows per point

    // build the gathered A panel (q | k[nbr] | edge | gene-diff | zero pad)
    for (int e = tid; e < 16 * KA_PAD; e += 128) {
        const int r = e / KA_PAD;
        const int f = e - r * KA_PAD;
        const int rem = (row0 + r) & 127;
        const int m = rem >> 3;
        const int h = rem & 7;
        _Float16 v = (_Float16)0.0f;
        if (f < 64)       v = qh[n * 512 + h * 64 + f];
        else if (f < 128) v = kh[(size_t)nbr[n * 16 + m] * 512 + h * 64 + (f - 64)];
        else if (f < 144) v = edgeh[(size_t)(n * 16 + m) * 128 + h * 16 + (f - 128)];
        else if (f < 194) v = gdh[(size_t)(n * 16 + m) * 50 + (f - 144)];
        Ap[r][f] = v;
    }

    const int wave = tid >> 5, lane = tid & 31, hi = lane >> 4, l16 = lane & 15;
    const v8f zero = {0.f, 0.f, 0.f, 0.f, 0.f, 0.f, 0.f, 0.f};
    v8f acc[8];
#pragma unroll
    for (int t = 0; t < 8; ++t) acc[t] = zero;

    __syncthreads();
    __builtin_prefetch(wa1t + (size_t)(wave << 7) * KA_PAD, 0, 0);
    for (int kc = 0; kc < 7; ++kc) {
        const int kb = kc << 5;
        const v16h a = cat16(*(const v8h*)&Ap[l16][kb + (hi << 3)],
                             *(const v8h*)&Ap[l16][kb + (hi << 3) + 16]);
#pragma unroll
        for (int t = 0; t < 8; ++t) {
            const int cl = (wave << 7) + (t << 4) + l16;
            const _Float16* bp = wa1t + (size_t)cl * KA_PAD + kb + (hi << 3);
            const v16h b = cat16(*(const v8h*)bp, *(const v8h*)(bp + 16));
            acc[t] = __builtin_amdgcn_wmma_f32_16x16x32_f16(false, a, false, b, (short)0,
                                                            acc[t], false, false);
        }
    }

    // bias + GELU into Hb
#pragma unroll
    for (int t = 0; t < 8; ++t) {
        const int cl = (wave << 7) + (t << 4) + l16;
        const float bv = ba1[cl];
#pragma unroll
        for (int r = 0; r < 8; ++r)
            Hb[r + (hi ? 8 : 0)][cl] = gelu_f(acc[t][r] + bv);
    }
    __syncthreads();

    // parallel per-row LN(512) + dot with wa2 -> logit (8 threads per row)
    const int r = tid >> 3;
    const int g = tid & 7;
    float s1 = 0.f, s2 = 0.f;
    for (int c = g; c < 512; c += 8) { const float v = Hb[r][c]; s1 += v; s2 += v * v; }
    red1[r][g] = s1; red2[r][g] = s2;
    __syncthreads();
    for (int s = 4; s > 0; s >>= 1) {
        if (g < s) { red1[r][g] += red1[r][g + s]; red2[r][g] += red2[r][g + s]; }
        __syncthreads();
    }
    const float mean = red1[r][0] * (1.f / 512.f);
    const float var  = red2[r][0] * (1.f / 512.f) - mean * mean;
    const float rstd = rsqrtf(var + 1e-5f);
    __syncthreads();
    float lg = 0.f;
    for (int c = g; c < 512; c += 8) lg += (Hb[r][c] - mean) * rstd * wa2[c];
    red1[r][g] = lg;
    __syncthreads();
    for (int s = 4; s > 0; s >>= 1) {
        if (g < s) red1[r][g] += red1[r][g + s];
        __syncthreads();
    }
    if (g == 0) logits[row0 + r] = red1[r][0] + ba2[0];
}

// ---------------------------------------------------------------------------
// Softmax over M=16 + context aggregation (attn@v, attn@edge) -> ctx_in [N,640] f16
// Block = (n, h), 64 threads.
// ---------------------------------------------------------------------------
__global__ __launch_bounds__(64) void attn_agg_kernel(
    const float* __restrict__ logits, const float* __restrict__ vf,
    const _Float16* __restrict__ edgeh, const int* __restrict__ nbr,
    _Float16* __restrict__ ctx_in)
{
    __shared__ float att[16];
    __shared__ int nb[16];
    const int n = blockIdx.x, h = blockIdx.y;
    const int t = threadIdx.x;
    if (t < 16) {
        att[t] = logits[(size_t)(n * 16 + t) * 8 + h];
        nb[t]  = nbr[n * 16 + t];
    }
    __syncthreads();
    if (t == 0) {
        float mx = att[0];
        for (int m = 1; m < 16; ++m) mx = fmaxf(mx, att[m]);
        float s = 0.f;
        for (int m = 0; m < 16; ++m) { att[m] = expf(att[m] - mx); s += att[m]; }
        const float inv = 1.f / s;
        for (int m = 0; m < 16; ++m) att[m] *= inv;
    }
    __syncthreads();
    float sc = 0.f;
#pragma unroll
    for (int m = 0; m < 16; ++m)
        sc += att[m] * vf[(size_t)nb[m] * 512 + h * 64 + t];
    ctx_in[(size_t)n * 640 + h * 64 + t] = (_Float16)sc;
    if (t < 16) {
        float ec = 0.f;
#pragma unroll
        for (int m = 0; m < 16; ++m)
            ec += att[m] * (float)edgeh[(size_t)(n * 16 + m) * 128 + h * 16 + t];
        ctx_in[(size_t)n * 640 + 512 + h * 16 + t] = (_Float16)ec;
    }
}

// ---------------------------------------------------------------------------
extern "C" void kernel_launch(void* const* d_in, const int* in_sizes, int n_in,
                              void* d_out, int out_size, void* d_ws, size_t ws_size,
                              hipStream_t stream)
{
    (void)in_sizes; (void)n_in; (void)out_size; (void)ws_size;

    const float* gene_exp = (const float*)d_in[0];
    const float* token    = (const float*)d_in[1];
    const float* coords   = (const float*)d_in[2];
    const int*   nbr      = (const int*)d_in[3];
    const float* w_qkv = (const float*)d_in[4];  const float* b_qkv = (const float*)d_in[5];
    const float* wa1 = (const float*)d_in[6];    const float* ba1 = (const float*)d_in[7];
    const float* wa2 = (const float*)d_in[8];    const float* ba2 = (const float*)d_in[9];
    const float* we1 = (const float*)d_in[10];   const float* be1 = (const float*)d_in[11];
    const float* we2 = (const float*)d_in[12];   const float* be2 = (const float*)d_in[13];
    const float* wo1 = (const float*)d_in[14];   const float* bo1 = (const float*)d_in[15];
    const float* wo2 = (const float*)d_in[16];   const float* bo2 = (const float*)d_in[17];
    const float* wm1 = (const float*)d_in[18];   const float* bm1 = (const float*)d_in[19];
    const float* wm2 = (const float*)d_in[20];   const float* bm2 = (const float*)d_in[21];
    const float* wg1 = (const float*)d_in[22];   const float* bg1 = (const float*)d_in[23];
    const float* wg2 = (const float*)d_in[24];   const float* bg2 = (const float*)d_in[25];

    float* out_gene = (float*)d_out;
    float* out_tok  = out_gene + (size_t)N_PTS * 50;

    // ---- workspace bump allocator (with lifetime-disjoint region reuse) ----
    char* ws = (char*)d_ws;
    size_t off = 0;
    auto alloc = [&](size_t bytes) -> char* {
        char* p = ws + off;
        off = (off + bytes + 255) & ~(size_t)255;
        return p;
    };

    // transposed f16 weights: [N][K]
    _Float16* wqkvt = (_Float16*)alloc(1536ull * 512 * 2);
    _Float16* wa1t  = (_Float16*)alloc(512ull * KA_PAD * 2);
    _Float16* we2t  = (_Float16*)alloc(128ull * 128 * 2);
    _Float16* wo1t  = (_Float16*)alloc(512ull * 640 * 2);
    _Float16* wo2t  = (_Float16*)alloc(512ull * 512 * 2);
    _Float16* wm1t  = (_Float16*)alloc(2048ull * 512 * 2);
    _Float16* wm2t  = (_Float16*)alloc(512ull * 2048 * 2);
    _Float16* wg1t  = (_Float16*)alloc(512ull * 512 * 2);
    _Float16* wg2t  = (_Float16*)alloc(128ull * 512 * 2);   // padded 50 -> 128 cols
    float*    bg2p  = (float*)alloc(128 * 4);

    _Float16* xnh = (_Float16*)alloc((size_t)N_PTS * 512 * 2);

    char* Rbig = alloc((size_t)N_PTS * 2048 * 4);   // 16MB: qkv f32 -> m1 f32 -> g1/g1h
    float*    qkvf = (float*)Rbig;
    float*    m1   = (float*)Rbig;
    float*    g1   = (float*)Rbig;
    _Float16* g1h  = (_Float16*)(Rbig + 8ull * 1024 * 1024);

    _Float16* qh  = (_Float16*)alloc((size_t)N_PTS * 512 * 2);
    _Float16* khb = (_Float16*)alloc((size_t)N_PTS * 512 * 2);
    float*    vf  = (float*)alloc((size_t)N_PTS * 512 * 4);
    float* proj   = (float*)alloc((size_t)N_PTS * 16 * 2 * 4);
    float* rnormb = (float*)alloc((size_t)N_PTS * 16 * 4);

    char* Rmid = alloc(32768ull * 128 * 2);         // 8MB: hn_avg -> m1h
    _Float16* hn_avg = (_Float16*)Rmid;
    _Float16* m1h    = (_Float16*)Rmid;

    _Float16* edgeh = (_Float16*)alloc(32768ull * 128 * 2);

    char* Rg = alloc(32768ull * 50 * 2);            // 3.3MB: gdh -> ctx_in
    _Float16* gdh    = (_Float16*)Rg;
    _Float16* ctx_in = (_Float16*)Rg;

    float* logitsb = (float*)alloc(32768ull * 8 * 4);

    char* Rt = alloc((size_t)N_PTS * 512 * 4);      // 4MB: t1 -> ctxf
    float* t1   = (float*)Rt;
    float* ctxf = (float*)Rt;

    _Float16* t1h   = (_Float16*)alloc((size_t)N_PTS * 512 * 2);
    float*    tok1  = (float*)alloc((size_t)N_PTS * 512 * 4);
    _Float16* tokh  = (_Float16*)alloc((size_t)N_PTS * 512 * 2);
    float*    m2    = (float*)alloc((size_t)N_PTS * 512 * 4);
    _Float16* tokfh = (_Float16*)alloc((size_t)N_PTS * 512 * 2);
    float*    g2    = (float*)alloc((size_t)N_PTS * 128 * 4); // padded gene out

    const int T = 256;
    auto cdiv = [](int a, int b) { return (a + b - 1) / b; };

    // ---- weight conversion: f32 [K][N] -> f16 transposed [N][K] (padded) ----
    convert_transpose_kernel<<<cdiv(1536 * 512, T), T, 0, stream>>>(w_qkv, wqkvt, 512, 1536, 512, 1536);
    convert_transpose_kernel<<<cdiv(512 * KA_PAD, T), T, 0, stream>>>(wa1, wa1t, 194, 512, KA_PAD, 512);
    convert_transpose_kernel<<<cdiv(128 * 128, T), T, 0, stream>>>(we2, we2t, 128, 128, 128, 128);
    convert_transpose_kernel<<<cdiv(512 * 640, T), T, 0, stream>>>(wo1, wo1t, 640, 512, 640, 512);
    convert_transpose_kernel<<<cdiv(512 * 512, T), T, 0, stream>>>(wo2, wo2t, 512, 512, 512, 512);
    convert_transpose_kernel<<<cdiv(2048 * 512, T), T, 0, stream>>>(wm1, wm1t, 512, 2048, 512, 2048);
    convert_transpose_kernel<<<cdiv(512 * 2048, T), T, 0, stream>>>(wm2, wm2t, 2048, 512, 2048, 512);
    convert_transpose_kernel<<<cdiv(512 * 512, T), T, 0, stream>>>(wg1, wg1t, 512, 512, 512, 512);
    convert_transpose_kernel<<<cdiv(128 * 512, T), T, 0, stream>>>(wg2, wg2t, 512, 50, 512, 128);
    pad_bias_kernel<<<1, 128, 0, stream>>>(bg2, bg2p, 50, 128);

    // ---- qkv = ln(token) @ w_qkv + b ----
    ln_f32_to_f16_kernel<<<N_PTS, 256, 0, stream>>>(token, xnh, 512);
    gemm_f16_kernel<<<dim3(N_PTS / 32, 12), 128, 0, stream>>>(
        xnh, wqkvt, b_qkv, qkvf, nullptr, 512, 1536, 0);
    split_qkv_kernel<<<cdiv(N_PTS * 512, T), T, 0, stream>>>(qkvf, qh, khb, vf);

    // ---- frame averaging + edge MLP ----
    frames_kernel<<<cdiv(N_PTS, T), T, 0, stream>>>(coords, nbr, proj, rnormb);
    edge_hidden_kernel<<<N_PTS * M_NBR, 128, 0, stream>>>(proj, rnormb, we1, be1, hn_avg);
    gemm_f16_kernel<<<dim3(N_PTS * M_NBR / 32, 1), 128, 0, stream>>>(
        hn_avg, we2t, be2, nullptr, edgeh, 128, 128, 0);

    // ---- per-edge MLP attention ----
    gd_kernel<<<cdiv(N_PTS * M_NBR * 50, T), T, 0, stream>>>(gene_exp, nbr, gdh);
    attn_fc_kernel<<<N_PTS * M_NBR * H_HEADS / 16, 128, 0, stream>>>(
        qh, khb, edgeh, gdh, nbr, wa1t, ba1, wa2, ba2, logitsb);
    attn_agg_kernel<<<dim3(N_PTS, H_HEADS), 64, 0, stream>>>(logitsb, vf, edgeh, nbr, ctx_in);

    // ---- ctx = mlp2([sc|ec], wo1, wo2) ; tok = token + ctx ----
    gemm_f16_kernel<<<dim3(N_PTS / 32, 4), 128, 0, stream>>>(
        ctx_in, wo1t, bo1, t1, nullptr, 640, 512, 1);
    ln_f32_to_f16_kernel<<<N_PTS, 256, 0, stream>>>(t1, t1h, 512);
    gemm_f16_kernel<<<dim3(N_PTS / 32, 4), 128, 0, stream>>>(
        t1h, wo2t, bo2, ctxf, nullptr, 512, 512, 0);
    add2_kernel<<<cdiv(N_PTS * 512, T), T, 0, stream>>>(token, ctxf, tok1, tokh, N_PTS * 512);

    // ---- tok = tok + mlp2(tok, wm1, wm2) (written straight into d_out tok slot) ----
    gemm_f16_kernel<<<dim3(N_PTS / 32, 16), 128, 0, stream>>>(
        tokh, wm1t, bm1, m1, nullptr, 512, 2048, 1);
    ln_f32_to_f16_kernel<<<N_PTS, 256, 0, stream>>>(m1, m1h, 2048);
    gemm_f16_kernel<<<dim3(N_PTS / 32, 4), 128, 0, stream>>>(
        m1h, wm2t, bm2, m2, nullptr, 2048, 512, 0);
    add2_kernel<<<cdiv(N_PTS * 512, T), T, 0, stream>>>(tok1, m2, out_tok, tokfh, N_PTS * 512);

    // ---- gene_out = mlp2(tok, wg1, wg2) (Nout padded 50 -> 128) ----
    gemm_f16_kernel<<<dim3(N_PTS / 32, 4), 128, 0, stream>>>(
        tokfh, wg1t, bg1, g1, nullptr, 512, 512, 1);
    ln_f32_to_f16_kernel<<<N_PTS, 256, 0, stream>>>(g1, g1h, 512);
    gemm_f16_kernel<<<dim3(N_PTS / 32, 1), 128, 0, stream>>>(
        g1h, wg2t, bg2p, g2, nullptr, 512, 128, 0);
    copy_out_kernel<<<cdiv(N_PTS * 50, T), T, 0, stream>>>(g2, out_gene);
}
